// MultiHeadAttention_64604898066548
// MI455X (gfx1250) — matrix operand order repaired and checked by reading, not compiled, not run on hardware
//
#include <hip/hip_runtime.h>
#include <hip/hip_bf16.h>
#include <stdint.h>

// ---------------------------------------------------------------------------
// MI455X / gfx1250 wave32 WMMA implementation of causal multi-head attention.
// All matmuls run on v_wmma_f32_16x16x32_bf16 (f32 accumulate).
// ---------------------------------------------------------------------------

#define BATCH   4
#define SEQ     2048
#define DMODEL  1024
#define NHEAD   16
#define HDIM    64
#define MTOT    (BATCH * SEQ)          // 8192 rows for the projection GEMMs

typedef __attribute__((ext_vector_type(16))) __bf16       v16bf;
typedef __attribute__((ext_vector_type(8)))  float        v8f;
typedef __attribute__((ext_vector_type(4)))  unsigned int u32x4;

union FragB {                     // 16x32 (A) or 32x16 (B) bf16 fragment: 8 VGPRs
  u32x4 u[2];
  v16bf v;
};
static_assert(sizeof(FragB) == 32, "fragment must be 8 VGPRs");

__device__ __forceinline__ unsigned short f2bf(float f) {
  unsigned int u = __float_as_uint(f);
  u += 0x7FFFu + ((u >> 16) & 1u);        // round-to-nearest-even
  return (unsigned short)(u >> 16);
}

__device__ __forceinline__ v8f wmma_bf16(v16bf a, v16bf b, v8f c) {
  // 8 args: (neg_a, A, neg_b, B, c_mod, C, reuse_a, reuse_b)
  return __builtin_amdgcn_wmma_f32_16x16x32_bf16(false, a, false, b,
                                                 (short)0, c, false, false);
}

// Row reduction across the 16 lanes that hold one output row (wave32:
// xor masks 1,2,4,8 stay inside each 16-lane half).
__device__ __forceinline__ float rowmax16(float x) {
#pragma unroll
  for (int m = 8; m >= 1; m >>= 1) x = fmaxf(x, __shfl_xor(x, m, 32));
  return x;
}
__device__ __forceinline__ float rowsum16(float x) {
#pragma unroll
  for (int m = 8; m >= 1; m >>= 1) x += __shfl_xor(x, m, 32);
  return x;
}

// ---------------------------------------------------------------------------
// fp32 -> bf16 conversion
// ---------------------------------------------------------------------------
__global__ __launch_bounds__(256) void k_f32_to_bf16(const float* __restrict__ src,
                                                     unsigned short* __restrict__ dst,
                                                     int n) {
  int i = blockIdx.x * 256 + threadIdx.x;
  if (i < n) dst[i] = f2bf(src[i]);
}

// ---------------------------------------------------------------------------
// C[M,N] = A[M,K](bf16) * W[N,K]^T(bf16) + bias[N]
// Block tile 128x128, BK=32, 256 threads = 8 waves in a 2x4 grid;
// each wave: 4 M-tiles x 2 N-tiles of 16x16 => 8 WMMA per K-step.
// LDS rows padded to 80B so fragment chunks are 16B aligned ds_load_b128.
// ---------------------------------------------------------------------------
template <bool OUT_F32>
__global__ __launch_bounds__(256) void k_gemm_bf16(const unsigned short* __restrict__ A,
                                                   const unsigned short* __restrict__ W,
                                                   const float* __restrict__ bias,
                                                   void* __restrict__ Cout,
                                                   int M, int N, int K) {
  __shared__ __align__(16) unsigned short As[128 * 40]; // 80B padded rows
  __shared__ __align__(16) unsigned short Bs[128 * 40];

  const int tid  = threadIdx.x;
  const int wid  = tid >> 5;
  const int lane = tid & 31;
  const int wr   = wid >> 2;     // 0..1  (64 rows each)
  const int wc   = wid & 3;      // 0..3  (32 cols each)
  const int hi   = lane >> 4;    // lane half
  const int lr   = lane & 15;

  const int bm = blockIdx.y * 128;
  const int bn = blockIdx.x * 128;

  v8f acc[4][2] = {};

  for (int k0 = 0; k0 < K; k0 += 32) {
    // ---- stage 128x32 bf16 tiles of A and W into LDS (16B chunks) ----
#pragma unroll
    for (int j = 0; j < 2; j++) {
      int ch  = tid + 256 * j;       // 512 chunks per tile
      int row = ch >> 2;             // 0..127
      int cc  = ch & 3;              // 0..3 (16B each = 8 bf16)
      const u32x4* ga = (const u32x4*)(A + (size_t)(bm + row) * K + k0 + cc * 8);
      const u32x4* gw = (const u32x4*)(W + (size_t)(bn + row) * K + k0 + cc * 8);
      *(u32x4*)((char*)As + row * 80 + cc * 16) = *ga;
      *(u32x4*)((char*)Bs + row * 80 + cc * 16) = *gw;
      if (k0 + 32 < K) {             // global_prefetch_b8 of next K-step
        __builtin_prefetch(A + (size_t)(bm + row) * K + (k0 + 32) + cc * 8, 0, 3);
        __builtin_prefetch(W + (size_t)(bn + row) * K + (k0 + 32) + cc * 8, 0, 3);
      }
    }
    __syncthreads();

    // ---- B fragments (32x16): lane = column, 16 contiguous K at hi*32B ----
    FragB bfr[2];
#pragma unroll
    for (int nt = 0; nt < 2; nt++) {
      const char* p = (const char*)Bs + (wc * 32 + nt * 16 + lr) * 80 + hi * 32;
      bfr[nt].u[0] = *(const u32x4*)p;
      bfr[nt].u[1] = *(const u32x4*)(p + 16);
    }
    // ---- A fragments (16x32): lane = row, 16 contiguous K at hi*32B ----
#pragma unroll
    for (int mt = 0; mt < 4; mt++) {
      FragB afr;
      const char* p = (const char*)As + (wr * 64 + mt * 16 + lr) * 80 + hi * 32;
      afr.u[0] = *(const u32x4*)p;
      afr.u[1] = *(const u32x4*)(p + 16);
#pragma unroll
      for (int nt = 0; nt < 2; nt++)
        acc[mt][nt] = wmma_bf16(afr.v, bfr[nt].v, acc[mt][nt]);
    }
    __syncthreads();
  }

  // ---- store: D layout row = i + 8*hi, col = lr ----
#pragma unroll
  for (int mt = 0; mt < 4; mt++)
#pragma unroll
    for (int nt = 0; nt < 2; nt++)
#pragma unroll
      for (int i = 0; i < 8; i++) {
        int row = bm + wr * 64 + mt * 16 + i + 8 * hi;
        int col = bn + wc * 32 + nt * 16 + lr;
        float val = acc[mt][nt][i] + bias[col];
        if constexpr (OUT_F32)
          ((float*)Cout)[(size_t)row * N + col] = val;
        else
          ((unsigned short*)Cout)[(size_t)row * N + col] = f2bf(val);
      }
}

// ---------------------------------------------------------------------------
// Flash attention: one WG per (batch, head, 128 query rows); each of the 8
// waves owns a 16-row q tile.  Per 32-key block: 4 WMMA for S = Q K^T,
// online softmax (shfl row-reduce + v_exp_f32), P staged bf16 through
// per-wave LDS (same-wave DS ops are in-order), 4 WMMA for O += P V.
// ---------------------------------------------------------------------------
__global__ __launch_bounds__(256) void k_attention(const unsigned short* __restrict__ Qb,
                                                   const unsigned short* __restrict__ Kb,
                                                   const unsigned short* __restrict__ Vb,
                                                   unsigned short* __restrict__ Ob) {
  __shared__ __align__(16) unsigned short Ks[32 * 72];      // 32 keys x 64 d, 144B rows
  __shared__ __align__(16) unsigned short Vt[64 * 40];      // transposed: 64 d x 32 keys, 80B rows
  __shared__ __align__(16) unsigned short Pls[8][16 * 40];  // per-wave P tile, 80B rows

  const int b  = blockIdx.z;
  const int h  = blockIdx.y;
  const int q0 = blockIdx.x * 128;

  const int tid  = threadIdx.x;
  const int wid  = tid >> 5;
  const int lane = tid & 31;
  const int hi   = lane >> 4;
  const int lr   = lane & 15;
  const int qt   = q0 + wid * 16;                 // wave's query tile base

  const size_t headoff = ((size_t)b * SEQ) * DMODEL + (size_t)h * HDIM;

  // ---- Q fragments held in registers (A layout, direct from global) ----
  FragB qf[2];
#pragma unroll
  for (int ds = 0; ds < 2; ds++) {
    const unsigned short* qrow =
        Qb + headoff + (size_t)(qt + lr) * DMODEL + ds * 32 + hi * 16;
    qf[ds].u[0] = *(const u32x4*)qrow;
    qf[ds].u[1] = *(const u32x4*)(qrow + 8);
  }

  v8f o[4] = {};
  float mrun[8], lrun[8];
#pragma unroll
  for (int i = 0; i < 8; i++) { mrun[i] = -1e30f; lrun[i] = 0.0f; }

  const int   nb    = (qt + 15) / 32 + 1;         // causal block count (wave)
  const int   nbmax = (q0 + 127) / 32 + 1;        // WG max (for barriers)
  const float scale = 0.125f;                     // 1/sqrt(64)

  for (int kb_i = 0; kb_i < nbmax; kb_i++) {
    const int kb = kb_i * 32;

    // ---- cooperative staging: K row-major, V transposed ----
    {
      int row = tid >> 3;                         // key 0..31
      int cc  = tid & 7;                          // 16B chunk 0..7
      const u32x4* gk = (const u32x4*)(Kb + headoff + (size_t)(kb + row) * DMODEL + cc * 8);
      *(u32x4*)((char*)Ks + row * 144 + cc * 16) = *gk;
      const unsigned short* vrow = Vb + headoff + (size_t)(kb + row) * DMODEL + cc * 8;
#pragma unroll
      for (int e = 0; e < 8; e++)                 // scatter V[key][d] -> Vt[d][key]
        *((unsigned short*)((char*)Vt + (cc * 8 + e) * 80) + row) = vrow[e];
    }
    __syncthreads();

    if (kb_i < nb) {
      // ---- S = Q K^T : two 16-key tiles, K-dim 64 split into 2x32 ----
      v8f s0 = {}, s1 = {};
#pragma unroll
      for (int ds = 0; ds < 2; ds++) {
        FragB kf0, kf1;
        const char* p0 = (const char*)Ks + lr * 144 + ds * 64 + hi * 32;
        const char* p1 = (const char*)Ks + (16 + lr) * 144 + ds * 64 + hi * 32;
        kf0.u[0] = *(const u32x4*)p0;  kf0.u[1] = *(const u32x4*)(p0 + 16);
        kf1.u[0] = *(const u32x4*)p1;  kf1.u[1] = *(const u32x4*)(p1 + 16);
        s0 = wmma_bf16(qf[ds].v, kf0.v, s0);
        s1 = wmma_bf16(qf[ds].v, kf1.v, s1);
      }

      // ---- online softmax (rows live at vgpr i + 8*hi, col = lr) ----
      unsigned short* pl = Pls[wid];
#pragma unroll
      for (int i = 0; i < 8; i++) {
        int   row = qt + i + 8 * hi;
        float a0  = (kb + lr      <= row) ? s0[i] * scale : -1e30f;
        float a1  = (kb + 16 + lr <= row) ? s1[i] * scale : -1e30f;
        float mb  = fmaxf(rowmax16(a0), rowmax16(a1));
        float mn  = fmaxf(mrun[i], mb);
        float p0v = __expf(a0 - mn);
        float p1v = __expf(a1 - mn);
        float cor = __expf(mrun[i] - mn);
        lrun[i] = lrun[i] * cor + rowsum16(p0v) + rowsum16(p1v);
        mrun[i] = mn;
#pragma unroll
        for (int dt = 0; dt < 4; dt++) o[dt][i] *= cor;   // rescale running O
        pl[(i + 8 * hi) * 40 + lr]      = f2bf(p0v);
        pl[(i + 8 * hi) * 40 + 16 + lr] = f2bf(p1v);
      }

      // ---- P fragment (A layout) from per-wave LDS; same-wave DS in-order ----
      FragB pf;
      const char* pp = (const char*)pl + lr * 80 + hi * 32;
      pf.u[0] = *(const u32x4*)pp;
      pf.u[1] = *(const u32x4*)(pp + 16);

      // ---- O += P V : B fragments from transposed V tile ----
#pragma unroll
      for (int dt = 0; dt < 4; dt++) {
        FragB vf;
        const char* vp = (const char*)Vt + (dt * 16 + lr) * 80 + hi * 32;
        vf.u[0] = *(const u32x4*)vp;
        vf.u[1] = *(const u32x4*)(vp + 16);
        o[dt] = wmma_bf16(pf.v, vf.v, o[dt]);
      }
    }
    __syncthreads();
  }

  // ---- normalize and store bf16 (merge-heads layout matches Q/K/V) ----
#pragma unroll
  for (int i = 0; i < 8; i++) {
    float inv = 1.0f / lrun[i];
    int   row = qt + i + 8 * hi;
#pragma unroll
    for (int dt = 0; dt < 4; dt++)
      Ob[headoff + (size_t)row * DMODEL + dt * 16 + lr] = f2bf(o[dt][i] * inv);
  }
}

// ---------------------------------------------------------------------------
// Host launcher
// ---------------------------------------------------------------------------
extern "C" void kernel_launch(void* const* d_in, const int* in_sizes, int n_in,
                              void* d_out, int out_size, void* d_ws, size_t ws_size,
                              hipStream_t stream) {
  (void)in_sizes; (void)n_in; (void)out_size;

  const float* X  = (const float*)d_in[0];
  const float* Wq = (const float*)d_in[1];
  const float* bq = (const float*)d_in[2];
  const float* Wk = (const float*)d_in[3];
  const float* bk = (const float*)d_in[4];
  const float* Wv = (const float*)d_in[5];
  const float* bv = (const float*)d_in[6];
  const float* Wo = (const float*)d_in[7];
  const float* bo = (const float*)d_in[8];

  const size_t xElems = (size_t)MTOT * DMODEL;    // 8192*1024
  const size_t wElems = (size_t)DMODEL * DMODEL;  // 1024*1024

  size_t need = (xElems * 2 /*Xb*/ + 4 * wElems * 2 /*weights*/ +
                 3 * xElems * 2 /*QKV*/ + xElems * 2 /*attn out*/);
  if (ws_size < need) return;                     // workspace too small: bail safely

  char* w = (char*)d_ws;
  unsigned short* Xb  = (unsigned short*)w;  w += xElems * 2;
  unsigned short* Wqb = (unsigned short*)w;  w += wElems * 2;
  unsigned short* Wkb = (unsigned short*)w;  w += wElems * 2;
  unsigned short* Wvb = (unsigned short*)w;  w += wElems * 2;
  unsigned short* Wob = (unsigned short*)w;  w += wElems * 2;
  unsigned short* Qb  = (unsigned short*)w;  w += xElems * 2;
  unsigned short* Kb  = (unsigned short*)w;  w += xElems * 2;
  unsigned short* Vb  = (unsigned short*)w;  w += xElems * 2;
  unsigned short* Ab  = (unsigned short*)w;  w += xElems * 2;

  // 1) fp32 -> bf16 conversions
  k_f32_to_bf16<<<dim3((unsigned)((xElems + 255) / 256)), 256, 0, stream>>>(X, Xb, (int)xElems);
  k_f32_to_bf16<<<dim3((unsigned)((wElems + 255) / 256)), 256, 0, stream>>>(Wq, Wqb, (int)wElems);
  k_f32_to_bf16<<<dim3((unsigned)((wElems + 255) / 256)), 256, 0, stream>>>(Wk, Wkb, (int)wElems);
  k_f32_to_bf16<<<dim3((unsigned)((wElems + 255) / 256)), 256, 0, stream>>>(Wv, Wvb, (int)wElems);
  k_f32_to_bf16<<<dim3((unsigned)((wElems + 255) / 256)), 256, 0, stream>>>(Wo, Wob, (int)wElems);

  // 2) Q/K/V projections: bf16 out
  dim3 ggrid(DMODEL / 128, MTOT / 128);           // (8, 64)
  k_gemm_bf16<false><<<ggrid, 256, 0, stream>>>(Xb, Wqb, bq, Qb, MTOT, DMODEL, DMODEL);
  k_gemm_bf16<false><<<ggrid, 256, 0, stream>>>(Xb, Wkb, bk, Kb, MTOT, DMODEL, DMODEL);
  k_gemm_bf16<false><<<ggrid, 256, 0, stream>>>(Xb, Wvb, bv, Vb, MTOT, DMODEL, DMODEL);

  // 3) causal flash attention
  dim3 agrid(SEQ / 128, NHEAD, BATCH);            // (16, 16, 4)
  k_attention<<<agrid, 256, 0, stream>>>(Qb, Kb, Vb, Ab);

  // 4) output projection: fp32 out + bo -> d_out
  k_gemm_bf16<true><<<ggrid, 256, 0, stream>>>(Ab, Wob, bo, d_out, MTOT, DMODEL, DMODEL);
}
